// MultiHeadAttention_82927228551772
// MI455X (gfx1250) — compile-verified, matrix operand order checked by reading, not run on hardware
//
#include <hip/hip_runtime.h>
#include <hip/hip_bf16.h>

// ---------------------------------------------------------------------------
// MultiHeadAttention forward for MI455X (gfx1250, wave32, WMMA).
//   D_MODEL=1024, HEAD=16, HEAD_DIM=64, BATCH=2, SEQ=2048
// Outputs (concatenated in d_out): z [B,S,D] fp32, attn [B,H,S,S] fp32.
// Intermediates (qp/kp/vp/z-head) live in f16 workspace: they are only ever
// consumed as WMMA f16 operands, so storing f32 would waste bandwidth + VALU.
// ---------------------------------------------------------------------------

#define D_MODEL 1024
#define NHEAD   16
#define HDIM    64
#define BATCH   2
#define SEQL    2048
#define MROWS   (BATCH * SEQL)   // 4096

typedef __attribute__((ext_vector_type(16))) _Float16 v16h;
typedef __attribute__((ext_vector_type(8)))  float    v8f;

// CDNA5 16-bit A/B 16x32 fragment: lane L (row/col L%16) holds
// K = {kb..kb+7, kb+16..kb+23}, kb = (L<16 ? 0 : 8); v16h elem i -> VGPR i/2.
__device__ __forceinline__ v16h frag_row_f32(const float* __restrict__ row, int kb) {
    v16h r;
#pragma unroll
    for (int j = 0; j < 8; ++j) {
        r[j]     = (_Float16)row[kb + j];
        r[8 + j] = (_Float16)row[kb + 16 + j];
    }
    return r;
}

// f16 source: two contiguous 16-byte chunks -> two global_load_b128, no cvt.
__device__ __forceinline__ v16h frag_row_f16(const _Float16* __restrict__ row, int kb) {
    v16h r;
#pragma unroll
    for (int j = 0; j < 8; ++j) {
        r[j]     = row[kb + j];
        r[8 + j] = row[kb + 16 + j];
    }
    return r;
}

__device__ __forceinline__ v8f wmma_f16(v16h a, v16h b, v8f c) {
    return __builtin_amdgcn_wmma_f32_16x16x32_f16(false, a, false, b, (short)0, c,
                                                  false, false);
}

// ---------------------------------------------------------------------------
// C[M,N] = X[M,K] @ W[N,K]^T + bias[N]      (torch Linear)
// XT: float (harness inputs) or _Float16 (f16 workspace).
// OutT: _Float16 (projection intermediates) or float (final output).
// Block: 128 threads = 4 waves; block tile 64x64; wave tile 16x64 with
// A-fragment reuse across 4 B fragments.
// ---------------------------------------------------------------------------
template <typename XT, typename OutT>
__global__ __launch_bounds__(128) void gemm_bias_nt(
    const XT* __restrict__ X, const float* __restrict__ W,
    const float* __restrict__ bias, OutT* __restrict__ C,
    int M, int N, int K) {
    const int lane = threadIdx.x & 31;
    const int wave = threadIdx.x >> 5;
    const int lm   = lane & 15;
    const int kb   = (lane >> 4) * 8;
    const int hfm  = (lane >> 4) * 8;      // row offset for C layout
    const int m0   = blockIdx.x * 64 + wave * 16;
    const int n0   = blockIdx.y * 64;

    v8f acc[4] = {};
    const XT* arow = X + (size_t)(m0 + lm) * K;

    for (int k = 0; k < K; k += 32) {
        __builtin_prefetch(arow + k + 128, 0, 1);     // global_prefetch_b8
        v16h a;
        if constexpr (sizeof(XT) == 2)
            a = frag_row_f16((const _Float16*)arow + k, kb);
        else
            a = frag_row_f32((const float*)arow + k, kb);
#pragma unroll
        for (int t = 0; t < 4; ++t) {
            const float* brow = W + (size_t)(n0 + t * 16 + lm) * K;
            v16h b = frag_row_f32(brow + k, kb);
            acc[t] = wmma_f16(a, b, acc[t]);
        }
    }
    // C/D layout: VGPR r -> (row r + hfm, col lm) per lane half.
#pragma unroll
    for (int t = 0; t < 4; ++t) {
        const int col = n0 + t * 16 + lm;
        const float bv = bias[col];
#pragma unroll
        for (int r = 0; r < 8; ++r) {
            C[(size_t)(m0 + r + hfm) * N + col] = (OutT)(acc[t][r] + bv);
        }
    }
}

// ---------------------------------------------------------------------------
// Fused attention for one (batch, head, 16-query block):
//   phase 1: S = (Qh * Kh^T) * scale  -> LDS (16 x 2048 fp32, 128 KB dyn-LDS)
//   phase 2: row softmax in LDS
//   phase 3: single coalesced write of attn to HBM (attn touches HBM once)
//   phase 4: z = P @ Vh; V tiles staged coalesced into LDS (4 KB), P re-read
//            from LDS with f32->f16 cvt, f32 accumulate, z stored f16.
// Block: 128 threads = 4 waves. grid = (SEQ/16, NHEAD, BATCH).
// ---------------------------------------------------------------------------
__global__ __launch_bounds__(128) void attn_fused(
    const _Float16* __restrict__ qp, const _Float16* __restrict__ kp,
    const _Float16* __restrict__ vp, float* __restrict__ attn,
    _Float16* __restrict__ zbuf) {
    extern __shared__ char smem[];
    float*    sc = (float*)smem;                               // [16][SEQL]
    _Float16* vt = (_Float16*)(smem + 16 * SEQL * sizeof(float)); // [32][64]
    __shared__ float red[16][8];

    const int b  = blockIdx.z;
    const int h  = blockIdx.y;
    const int q0 = blockIdx.x * 16;
    const int lane = threadIdx.x & 31;
    const int wave = threadIdx.x >> 5;
    const int lm   = lane & 15;
    const int kb   = (lane >> 4) * 8;
    const int hfm  = (lane >> 4) * 8;
    const float scale = 0.125f;            // 1/sqrt(64)

    // --- Q fragments for d=0..31 and d=32..63 (held for the whole kernel) ---
    const _Float16* qrow =
        qp + (size_t)(b * SEQL + q0 + lm) * D_MODEL + h * HDIM;
    const v16h aq0 = frag_row_f16(qrow, kb);
    const v16h aq1 = frag_row_f16(qrow + 32, kb);

    // --- phase 1: each wave covers 512 key columns (32 tiles of 16) ---------
    for (int t = 0; t < 32; ++t) {
        const int n0 = wave * 512 + t * 16;
        const _Float16* krow =
            kp + (size_t)(b * SEQL + n0 + lm) * D_MODEL + h * HDIM;
        __builtin_prefetch(krow + 16 * D_MODEL, 0, 1);
        v16h bk0 = frag_row_f16(krow, kb);
        v16h bk1 = frag_row_f16(krow + 32, kb);
        v8f acc = {};
        acc = wmma_f16(aq0, bk0, acc);
        acc = wmma_f16(aq1, bk1, acc);
#pragma unroll
        for (int r = 0; r < 8; ++r)
            sc[(size_t)(r + hfm) * SEQL + n0 + lm] = acc[r] * scale;
    }
    __syncthreads();

    // --- phase 2: softmax; 8 threads per row, 256 cols each -----------------
    {
        const int row = threadIdx.x >> 3;
        const int sub = threadIdx.x & 7;
        float* rp = sc + (size_t)row * SEQL;
        float mx = -3.0e38f;
        for (int c = sub * 256; c < sub * 256 + 256; ++c) mx = fmaxf(mx, rp[c]);
        red[row][sub] = mx;
        __syncthreads();
        mx = red[row][0];
#pragma unroll
        for (int i = 1; i < 8; ++i) mx = fmaxf(mx, red[row][i]);
        __syncthreads();
        float s = 0.0f;
        for (int c = sub * 256; c < sub * 256 + 256; ++c) {
            float e = __expf(rp[c] - mx);
            rp[c] = e;
            s += e;
        }
        red[row][sub] = s;
        __syncthreads();
        s = 0.0f;
#pragma unroll
        for (int i = 0; i < 8; ++i) s += red[row][i];
        const float inv = 1.0f / s;
        for (int c = sub * 256; c < sub * 256 + 256; ++c) rp[c] *= inv;
    }
    __syncthreads();

    // --- phase 3: coalesced attn write (row stride SEQL matches LDS flat) ---
    {
        float* aout = attn + ((size_t)(b * NHEAD + h) * SEQL + q0) * SEQL;
        for (int i = threadIdx.x; i < 16 * SEQL; i += 128) aout[i] = sc[i];
    }

    // --- phase 4: z = P @ V; each wave owns one 16-col slice of head dim ----
    {
        v8f acc = {};
        const int n0 = wave * 16;
        const float* srow = sc + (size_t)lm * SEQL;   // A row = lm (both halves)
        // Cooperative V staging: thread t copies 16 contiguous halves (32 B)
        // of the 32x64 tile; source rows are contiguous in vp's head slice.
        const int vrow = threadIdx.x >> 2;            // 0..31
        const int vcol = (threadIdx.x & 3) * 16;      // 0,16,32,48
        for (int kt = 0; kt < SEQL; kt += 32) {
            __syncthreads();                          // vt free from last iter
            const _Float16* vsrc =
                vp + (size_t)(b * SEQL + kt + vrow) * D_MODEL + h * HDIM + vcol;
#pragma unroll
            for (int j = 0; j < 16; ++j) vt[vrow * HDIM + vcol + j] = vsrc[j];
            __syncthreads();
            // A fragment: probs from LDS (f32 -> f16)
            v16h a;
#pragma unroll
            for (int j = 0; j < 8; ++j) {
                a[j]     = (_Float16)srow[kt + kb + j];
                a[8 + j] = (_Float16)srow[kt + kb + 16 + j];
            }
            // B fragment: column (n0+lm) of the staged V tile; lanes hit
            // consecutive LDS banks (2-byte elems, consecutive columns).
            v16h bv;
#pragma unroll
            for (int j = 0; j < 8; ++j) {
                bv[j]     = vt[(kb + j) * HDIM + n0 + lm];
                bv[8 + j] = vt[(kb + 16 + j) * HDIM + n0 + lm];
            }
            acc = wmma_f16(a, bv, acc);
        }
        _Float16* zrow =
            zbuf + (size_t)(b * SEQL + q0) * D_MODEL + h * HDIM + n0;
#pragma unroll
        for (int r = 0; r < 8; ++r)
            zrow[(size_t)(r + hfm) * D_MODEL + lm] = (_Float16)acc[r];
    }
}

// ---------------------------------------------------------------------------
extern "C" void kernel_launch(void* const* d_in, const int* in_sizes, int n_in,
                              void* d_out, int out_size, void* d_ws, size_t ws_size,
                              hipStream_t stream) {
    (void)in_sizes; (void)n_in; (void)out_size; (void)ws_size;
    const float* q   = (const float*)d_in[0];
    const float* k   = (const float*)d_in[1];
    const float* v   = (const float*)d_in[2];
    const float* w_q = (const float*)d_in[3];
    const float* b_q = (const float*)d_in[4];
    const float* w_k = (const float*)d_in[5];
    const float* b_k = (const float*)d_in[6];
    const float* w_v = (const float*)d_in[7];
    const float* b_v = (const float*)d_in[8];
    const float* w_o = (const float*)d_in[9];
    const float* b_o = (const float*)d_in[10];

    float* out_z    = (float*)d_out;                               // [B,S,D]
    float* out_attn = out_z + (size_t)MROWS * D_MODEL;             // [B,H,S,S]

    const size_t chunk = (size_t)MROWS * D_MODEL;                  // elems
    _Float16* qp   = (_Float16*)d_ws;
    _Float16* kp   = qp + chunk;
    _Float16* vp   = kp + chunk;
    _Float16* zbuf = vp + chunk;

    dim3 gblk(128);
    dim3 ggrid(MROWS / 64, D_MODEL / 64);                          // (64,16)
    hipLaunchKernelGGL((gemm_bias_nt<float, _Float16>), ggrid, gblk, 0, stream,
                       q, w_q, b_q, qp, MROWS, D_MODEL, D_MODEL);
    hipLaunchKernelGGL((gemm_bias_nt<float, _Float16>), ggrid, gblk, 0, stream,
                       k, w_k, b_k, kp, MROWS, D_MODEL, D_MODEL);
    hipLaunchKernelGGL((gemm_bias_nt<float, _Float16>), ggrid, gblk, 0, stream,
                       v, w_v, b_v, vp, MROWS, D_MODEL, D_MODEL);

    dim3 agrid(SEQL / 16, NHEAD, BATCH);                           // (128,16,2)
    const size_t lds_bytes =
        (size_t)16 * SEQL * sizeof(float) + (size_t)32 * HDIM * sizeof(_Float16);
    hipLaunchKernelGGL(attn_fused, agrid, dim3(128), lds_bytes, stream,
                       qp, kp, vp, out_attn, zbuf);

    hipLaunchKernelGGL((gemm_bias_nt<_Float16, float>), ggrid, gblk, 0, stream,
                       zbuf, w_o, b_o, out_z, MROWS, D_MODEL, D_MODEL);
}
